// SympleAgent_36661840839777
// MI455X (gfx1250) — matrix operand order repaired and checked by reading, not compiled, not run on hardware
//
#include <hip/hip_runtime.h>
#include <hip/hip_bf16.h>
#include <math.h>

// ---------- CDNA5 WMMA types ----------
typedef __attribute__((ext_vector_type(16))) __bf16 v16bf;
typedef __attribute__((ext_vector_type(8)))  float  v8f;

union Frag {
    uint4 q[2];   // 32 bytes = 16 bf16
    v16bf v;
};

// ---------- problem constants ----------
constexpr int kH      = 512;
constexpr int kE      = 512;
constexpr int kN      = 1023;
constexpr int kB      = 64;
constexpr int kOPS    = 16;
constexpr int kGCOLS  = 2048;           // 4 gates * H, gate-interleaved: col = 4*h + gate
constexpr int kKTOT   = 1536;           // E + H(left) + H(right)
constexpr int TILE_M  = 128;
constexpr int TILE_N  = 128;
constexpr int TILE_K  = 32;
constexpr int KSTEPS  = kKTOT / TILE_K; // 48
constexpr int LDA     = 40;             // u16 pitch for A/B LDS staging (80B, 16B-aligned rows)
constexpr int LDG     = 132;            // f32 pitch for gate tile in LDS
constexpr int STAGE_BYTES = TILE_M * LDA * 2;      // 10240 per A or B tile
constexpr int BUF_BYTES   = 2 * STAGE_BYTES;       // A+B per buffer = 20480

static __device__ __forceinline__ unsigned short f2bf(float f) {
    unsigned int u = __float_as_uint(f);
    unsigned int r = u + 0x7FFFu + ((u >> 16) & 1u);   // round-to-nearest-even
    return (unsigned short)(r >> 16);
}

// ---------- weight pre-pack: Wcat[2048][1536] bf16 + fused bias[2048] ----------
__global__ void prep_weights_kernel(const float* __restrict__ W_ih,
                                    const float* __restrict__ W_hh,
                                    const float* __restrict__ b_ih,
                                    const float* __restrict__ b_hh,
                                    unsigned short* __restrict__ wcat,
                                    float* __restrict__ biasr) {
    const long total = (long)kGCOLS * kKTOT;
    for (long i = (long)blockIdx.x * blockDim.x + threadIdx.x; i < total;
         i += (long)gridDim.x * blockDim.x) {
        int col  = (int)(i / kKTOT);
        int k    = (int)(i - (long)col * kKTOT);
        int gate = col & 3;
        int h    = col >> 2;
        int orow = gate * 1024 + h;     // only rows [gate*2H, gate*2H + H) contribute
        float w;
        if (k < kE)            w = W_ih[(long)orow * kE + k];
        else if (k < kE + kH)  w = W_hh[(long)orow * (2 * kH) + (k - kE)];
        else                   w = W_hh[(long)orow * (2 * kH) + kH + (k - kE - kH)];
        wcat[i] = f2bf(w);
        if (k == 0) biasr[col] = b_ih[orow] + b_hh[orow];
    }
}

// ---------- embedding gather (+ int-arg override on last channel), f32 -> bf16 ----------
__global__ void prep_emb_kernel(const int* __restrict__ node_types,
                                const float* __restrict__ node_args,
                                const float* __restrict__ emb_table,
                                unsigned short* __restrict__ embbf) {
    const long total = (long)kB * kN * kE;
    for (long i = (long)blockIdx.x * blockDim.x + threadIdx.x; i < total;
         i += (long)gridDim.x * blockDim.x) {
        int  e  = (int)(i & (kE - 1));
        long bn = i >> 9;               // b*kN + n
        int  t  = node_types[bn];
        float v = emb_table[(long)t * kE + e];
        if (e == kE - 1 && t <= 1) v = node_args[bn];
        embbf[i] = f2bf(v);
    }
}

// ---------- zero h/c at the 512 leaf nodes (511..1022) ----------
__global__ void zero_leaves_kernel(unsigned short* __restrict__ hbf,
                                   float* __restrict__ cbuf) {
    const long total = (long)kB * 512 * kH;
    for (long i = (long)blockIdx.x * blockDim.x + threadIdx.x; i < total;
         i += (long)gridDim.x * blockDim.x) {
        int  e    = (int)(i & 511);
        long r    = i >> 9;
        int  b    = (int)(r >> 9);
        int  leaf = (int)(r & 511) + 511;
        long off  = (((long)b * kN + leaf) << 9) + e;
        hbf[off]  = 0;
        cbuf[off] = 0.0f;
    }
}

// ---------- fused per-level GEMM (bf16 WMMA, async-DMA staged) + LSTM cell ----------
__global__ __launch_bounds__(256)
void tree_level_kernel(const unsigned short* __restrict__ embbf,
                       unsigned short* __restrict__ hbf,
                       float* __restrict__ cbuf,
                       const unsigned short* __restrict__ wcat,
                       const float* __restrict__ biasr,
                       float* __restrict__ hroot,
                       int dshift, int first, int Mrows, int isRoot) {
    // 66KB: gate tile aliases the two double-buffered staging tiles (4 x 10KB)
    __shared__ __align__(16) unsigned char smem_raw[TILE_M * LDG * 4];

    const int t     = threadIdx.x;
    const int lane  = t & 31;
    const int wave  = t >> 5;                 // 8 waves
    const int wr    = (wave & 3) * 32;        // wave row offset (2x16 sub-rows)
    const int wc    = (wave >> 2) * 64;       // wave col offset (4x16 sub-cols)
    const int mtile = blockIdx.x;
    const int ntile = blockIdx.y;
    const int n0    = ntile * TILE_N;
    const int nmask = (1 << dshift) - 1;

    // Async LDS DMA staging of one K-tile into buffer bufSel.
    // Each thread copies 2x16B of A (EXEC-masked for rows >= Mrows; stale LDS
    // garbage only reaches D-rows that are never stored) and 2x16B of B.
    auto stage_async = [&](int k0, int bufSel) {
#pragma unroll
        for (int qi = 0; qi < 2; ++qi) {
            int q    = t * 2 + qi;            // 0..511
            int row  = q >> 2;
            int col8 = (q & 3) * 8;           // u16 offset, 16B granules
            unsigned aOff = (unsigned)(bufSel * BUF_BYTES + (row * LDA + col8) * 2);
            unsigned bOff = aOff + (unsigned)STAGE_BYTES;
            int m = mtile * TILE_M + row;
            if (m < Mrows) {
                int b    = m >> dshift;
                int node = first + (m & nmask);
                int kg   = k0 + col8;
                int src  = kg >> 9;           // 0: emb, 1: h_left, 2: h_right
                int ks   = kg & 511;
                const unsigned short* ap;
                if (src == 0)      ap = embbf + ((((long)b * kN + node)         << 9) + ks);
                else if (src == 1) ap = hbf   + ((((long)b * kN + 2 * node + 1) << 9) + ks);
                else               ap = hbf   + ((((long)b * kN + 2 * node + 2) << 9) + ks);
                asm volatile("global_load_async_to_lds_b128 %0, %1, off"
                             :: "v"(aOff), "v"((unsigned long long)(size_t)ap)
                             : "memory");
            }
            const unsigned short* bp = wcat + ((long)(n0 + row) * kKTOT + k0 + col8);
            asm volatile("global_load_async_to_lds_b128 %0, %1, off"
                         :: "v"(bOff), "v"((unsigned long long)(size_t)bp)
                         : "memory");
            if (k0 + 2 * TILE_K < kKTOT)
                __builtin_prefetch(bp + 2 * TILE_K, 0, 0);   // global_prefetch_b8
        }
    };

    v8f acc[2][4] = {};

    stage_async(0, 0);                         // prime the pipeline
    for (int kb = 0; kb < KSTEPS; ++kb) {
        const int cur = kb & 1;
        // Own DMA done (ASYNCcnt), then cross-wave visibility (barrier).
        asm volatile("s_wait_asynccnt 0x0" ::: "memory");
        __syncthreads();
        // Overlap next tile's DMA with this tile's WMMAs (other buffer; all
        // waves finished reading it before the barrier above).
        if (kb + 1 < KSTEPS) stage_async((kb + 1) * TILE_K, 1 - cur);

        const unsigned short* Asb = (const unsigned short*)(smem_raw + cur * BUF_BYTES);
        const unsigned short* Bsb = Asb + TILE_M * LDA;

        // ---- 8 WMMAs per wave per K-step ----
#pragma unroll
        for (int mi = 0; mi < 2; ++mi) {
            Frag a;
            int lm = wr + mi * 16 + (lane & 15);
            int kbh = (lane & 16) >> 1;       // A lanes 16-31 hold K+8 (ISA 16-bit A layout)
            a.q[0] = *(const uint4*)&Asb[lm * LDA + kbh];
            a.q[1] = *(const uint4*)&Asb[lm * LDA + 16 + kbh];
#pragma unroll
            for (int ni = 0; ni < 4; ++ni) {
                Frag bf_;
                int ln = wc + ni * 16 + (lane & 15);
                int ko = (lane & 16) >> 1;    // B lanes 16-31 hold K+8
                bf_.q[0] = *(const uint4*)&Bsb[ln * LDA + ko];
                bf_.q[1] = *(const uint4*)&Bsb[ln * LDA + 16 + ko];
                acc[mi][ni] = __builtin_amdgcn_wmma_f32_16x16x32_bf16(
                    false, a.v, false, bf_.v, (short)0, acc[mi][ni], false, false);
            }
        }
    }
    __syncthreads();   // all frag reads done before gate tile overwrites staging

    // ---- dump accumulators to LDS (C/D layout: VGPR v -> M = v + 8*(lane>=16)) ----
    float* gsm = (float*)smem_raw;
#pragma unroll
    for (int mi = 0; mi < 2; ++mi)
#pragma unroll
        for (int ni = 0; ni < 4; ++ni)
#pragma unroll
            for (int v = 0; v < 8; ++v) {
                int r = wr + mi * 16 + v + ((lane & 16) >> 1);
                int c = wc + ni * 16 + (lane & 15);
                gsm[r * LDG + c] = acc[mi][ni][v];
            }
    __syncthreads();

    // ---- fused LSTM cell: cols 4u..4u+3 are (i,f,g,o) of hidden unit ntile*32+u ----
    const int hbase = ntile * 32;
#pragma unroll 4
    for (int it = 0; it < 16; ++it) {
        int p   = t + 256 * it;               // 128 rows x 32 hidden units
        int row = p >> 5;
        int u   = p & 31;
        int m   = mtile * TILE_M + row;
        if (m >= Mrows) continue;
        int b    = m >> dshift;
        int node = first + (m & nmask);
        int hidx = hbase + u;
        int c4   = 4 * u;
        float ig = gsm[row * LDG + c4 + 0] + biasr[n0 + c4 + 0];
        float fg = gsm[row * LDG + c4 + 1] + biasr[n0 + c4 + 1];
        float gg = gsm[row * LDG + c4 + 2] + biasr[n0 + c4 + 2];
        float og = gsm[row * LDG + c4 + 3] + biasr[n0 + c4 + 3];
        float cl = cbuf[(((long)b * kN + 2 * node + 1) << 9) + hidx]; // c0[:,:H] == c[left]
        float i_s = 1.0f / (1.0f + __expf(-ig));
        float f_s = 1.0f / (1.0f + __expf(-fg));
        float o_s = 1.0f / (1.0f + __expf(-og));
        float g_t = tanhf(gg);
        float cn  = f_s * cl + i_s * g_t;
        float hn  = o_s * tanhf(cn);
        long ofs  = (((long)b * kN + node) << 9) + hidx;
        cbuf[ofs] = cn;
        hbf[ofs]  = f2bf(hn);
        if (isRoot) hroot[((long)b << 9) + hidx] = hn;
    }
}

// ---------- tiny head: relu(h@W1^T+b1) @ W2^T + b2 + log(valid), softmax(T=3) ----------
__global__ __launch_bounds__(256)
void mlp_softmax_kernel(const float* __restrict__ hroot,
                        const float* __restrict__ W1, const float* __restrict__ b1,
                        const float* __restrict__ W2, const float* __restrict__ b2,
                        const float* __restrict__ validity,
                        float* __restrict__ out) {
    __shared__ float hs[kH];
    __shared__ float xs[kH];
    __shared__ float lg[kOPS];
    int b = blockIdx.x;
    int t = threadIdx.x;
    hs[t]       = hroot[(long)b * kH + t];
    hs[t + 256] = hroot[(long)b * kH + t + 256];
    __syncthreads();
    for (int o = t; o < kH; o += 256) {
        float acc = b1[o];
        const float* wr = W1 + (long)o * kH;
        for (int k = 0; k < kH; ++k) acc += hs[k] * wr[k];
        xs[o] = fmaxf(acc, 0.0f);
    }
    __syncthreads();
    if (t < kOPS) {
        float acc = b2[t];
        const float* wr = W2 + (long)t * kH;
        for (int k = 0; k < kH; ++k) acc += xs[k] * wr[k];
        lg[t] = acc + __logf(validity[(long)b * kOPS + t]);
    }
    __syncthreads();
    if (t < kOPS) {
        float mx = -1e30f;
        for (int o = 0; o < kOPS; ++o) mx = fmaxf(mx, lg[o]);
        float s = 0.0f;
        for (int o = 0; o < kOPS; ++o) s += __expf((lg[o] - mx) * (1.0f / 3.0f));
        out[(long)b * kOPS + t] = __expf((lg[t] - mx) * (1.0f / 3.0f)) / s;
    }
}

extern "C" void kernel_launch(void* const* d_in, const int* in_sizes, int n_in,
                              void* d_out, int out_size, void* d_ws, size_t ws_size,
                              hipStream_t stream) {
    const int*   node_types = (const int*)d_in[0];
    const float* node_args  = (const float*)d_in[1];
    const float* validity   = (const float*)d_in[2];
    const float* emb_table  = (const float*)d_in[3];
    const float* W_ih       = (const float*)d_in[4];
    const float* W_hh       = (const float*)d_in[5];
    const float* b_ih       = (const float*)d_in[6];
    const float* b_hh       = (const float*)d_in[7];
    const float* W1         = (const float*)d_in[8];
    const float* b1         = (const float*)d_in[9];
    const float* W2         = (const float*)d_in[10];
    const float* b2         = (const float*)d_in[11];
    float* out = (float*)d_out;

    char* p = (char*)d_ws;
    auto take = [&](size_t bytes) -> char* {
        char* r = p;
        p += (bytes + 255) & ~(size_t)255;
        return r;
    };
    unsigned short* embbf = (unsigned short*)take((size_t)kB * kN * kE * 2);
    unsigned short* hbf   = (unsigned short*)take((size_t)kB * kN * kH * 2);
    float*          cbuf  = (float*)take((size_t)kB * kN * kH * 4);
    unsigned short* wcat  = (unsigned short*)take((size_t)kGCOLS * kKTOT * 2);
    float*          biasr = (float*)take((size_t)kGCOLS * 4);
    float*          hroot = (float*)take((size_t)kB * kH * 4);

    prep_weights_kernel<<<2048, 256, 0, stream>>>(W_ih, W_hh, b_ih, b_hh, wcat, biasr);
    prep_emb_kernel<<<8192, 256, 0, stream>>>(node_types, node_args, emb_table, embbf);
    zero_leaves_kernel<<<4096, 256, 0, stream>>>(hbf, cbuf);

    for (int d = 8; d >= 0; --d) {
        int nodes  = 1 << d;
        int first  = nodes - 1;
        int Mrows  = kB * nodes;
        int mtiles = (Mrows + TILE_M - 1) / TILE_M;
        tree_level_kernel<<<dim3(mtiles, kGCOLS / TILE_N), 256, 0, stream>>>(
            embbf, hbf, cbuf, wcat, biasr, hroot, d, first, Mrows, (d == 0) ? 1 : 0);
    }
    mlp_softmax_kernel<<<kB, 256, 0, stream>>>(hroot, W1, b1, W2, b2, validity, out);
}
